// GNNEncoder_80659485819622
// MI455X (gfx1250) — matrix operand order repaired and checked by reading, not compiled, not run on hardware
//
#include <hip/hip_runtime.h>
#include <hip/hip_bf16.h>

// ---------- WMMA types ----------
typedef __attribute__((ext_vector_type(16))) __bf16 v16bf;
typedef __attribute__((ext_vector_type(8)))  float  v8f;

union B16 {
  v16bf v;
  unsigned short s[16];
  unsigned int   u[8];
  uint4          q[2];
};

__device__ inline unsigned short f2bf(float f) {
  unsigned int u = __float_as_uint(f);
  u += 0x7FFFu + ((u >> 16) & 1u);           // round-to-nearest-even
  return (unsigned short)(u >> 16);
}

__device__ inline v8f wmma_bf16(v16bf a, v16bf b, v8f c) {
  // (neg_a, A, neg_b, B, c_mod, C, reuse_a, reuse_b)
  return __builtin_amdgcn_wmma_f32_16x16x32_bf16(false, a, false, b, (short)0, c,
                                                 false, false);
}

#define NB     4
#define NNODES 2048
#define NHEAD  4
#define DK     64
#define FIN    128
#define HID    256
#define ALPHA  0.2f
#define NEGV   (-1.0e9f)

// =====================================================================
// Conversion kernels (bf16, WMMA-friendly layouts)
// =====================================================================
__global__ __launch_bounds__(256) void f32_to_bf16_kernel(
    const float* __restrict__ in, unsigned int* __restrict__ out, int npairs)
{
  int i = blockIdx.x * 256 + threadIdx.x;
  if (i >= npairs) return;
  float f0 = in[2 * i], f1 = in[2 * i + 1];
  out[i] = ((unsigned int)f2bf(f1) << 16) | (unsigned int)f2bf(f0);
}

// W (H,Kdim,Ncol) f32 -> Wt (H,Ncol,Kdim) bf16
__global__ __launch_bounds__(256) void w_to_bf16t_kernel(
    const float* __restrict__ W, unsigned short* __restrict__ Wt,
    int Kdim, int Ncol)
{
  int head = blockIdx.y;
  int idx  = blockIdx.x * 256 + threadIdx.x;
  if (idx >= Kdim * Ncol) return;
  int col = idx / Kdim, k = idx % Kdim;
  Wt[((size_t)head * Ncol + col) * Kdim + k] =
      f2bf(W[((size_t)head * Kdim + k) * Ncol + col]);
}

// h (B,H,N,DK) f32 -> hT (B,H,DK,N) bf16
__global__ __launch_bounds__(256) void h_to_bf16t_kernel(
    const float* __restrict__ h, unsigned short* __restrict__ hT)
{
  size_t i = (size_t)blockIdx.x * 256 + threadIdx.x;   // B*H*N*DK
  int d = (int)(i & (DK - 1));
  size_t bhn = i >> 6;
  int n = (int)(bhn & (NNODES - 1));
  size_t bh = bhn >> 11;
  hT[(bh * DK + d) * NNODES + n] = f2bf(h[i]);
}

// =====================================================================
// Per-head GEMM:  out[b,head,n,:] = X[row,:] @ W[head,:,:] (+bias)
// Each wave computes 32(M) x 64(N); K-loop fully unrolled.
// grid: (R/128, H, NC/64), block: 128 (4 independent waves)
// =====================================================================
template <int KD, int NC>
__global__ __launch_bounds__(128) void gemm_head_kernel(
    const unsigned short* __restrict__ Xbf,
    const unsigned short* __restrict__ Wt,
    const float* __restrict__ bias, float* __restrict__ out)
{
  const int tid  = threadIdx.x;
  const int wave = tid >> 5;
  const int lane = tid & 31;
  const int half = lane >> 4;
  const int l16  = lane & 15;
  const int rowBase = (blockIdx.x * 4 + wave) * 32;
  const int head    = blockIdx.y;
  const int Hh      = gridDim.y;
  const int n0      = blockIdx.z * 64;

  const unsigned short* xrow0 = Xbf + (size_t)(rowBase + l16) * KD;
  const unsigned short* xrow1 = xrow0 + (size_t)16 * KD;
  const unsigned short* wbase = Wt + ((size_t)head * NC + n0) * KD;

  v8f acc[2][4];
  for (int t = 0; t < 2; ++t)
    for (int s = 0; s < 4; ++s)
      for (int v = 0; v < 8; ++v) acc[t][s][v] = 0.0f;

#pragma unroll
  for (int k0 = 0; k0 < KD; k0 += 32) {
    B16 A0, A1, Bv[4];
    A0.q[0] = *(const uint4*)(xrow0 + k0 + 8 * half);
    A0.q[1] = *(const uint4*)(xrow0 + k0 + 16 + 8 * half);
    A1.q[0] = *(const uint4*)(xrow1 + k0 + 8 * half);
    A1.q[1] = *(const uint4*)(xrow1 + k0 + 16 + 8 * half);
#pragma unroll
    for (int s = 0; s < 4; ++s) {
      const unsigned short* wcol =
          wbase + (size_t)(s * 16 + l16) * KD + k0 + 8 * half;
      Bv[s].q[0] = *(const uint4*)(wcol);
      Bv[s].q[1] = *(const uint4*)(wcol + 16);
    }
#pragma unroll
    for (int s = 0; s < 4; ++s) {
      acc[0][s] = wmma_bf16(A0.v, Bv[s].v, acc[0][s]);
      acc[1][s] = wmma_bf16(A1.v, Bv[s].v, acc[1][s]);
    }
  }

  for (int t = 0; t < 2; ++t) {
    for (int v = 0; v < 8; ++v) {
      const int row = rowBase + t * 16 + v + 8 * half;  // C: M = v + 8*half
      const int b = row / NNODES, n = row % NNODES;
      for (int s = 0; s < 4; ++s) {
        const int col = n0 + s * 16 + l16;
        float val = acc[t][s][v] + (bias ? bias[col] : 0.0f);
        out[(((size_t)b * Hh + head) * NNODES + n) * NC + col] = val;
      }
    }
  }
}

// =====================================================================
// src/dst attention logits
// =====================================================================
__global__ __launch_bounds__(256) void srcdst_kernel(
    const float* __restrict__ h, const float* __restrict__ a,
    float* __restrict__ srcb, float* __restrict__ dstb)
{
  int idx = blockIdx.x * blockDim.x + threadIdx.x;   // B*H*N
  if (idx >= NB * NHEAD * NNODES) return;
  int head = (idx / NNODES) & (NHEAD - 1);
  const float* hv = h + (size_t)idx * DK;
  const float* as = a + head * 2 * DK;
  const float* ad = as + DK;
  float s = 0.0f, d = 0.0f;
  for (int i = 0; i < DK; ++i) {
    float x = hv[i];
    s += x * as[i];
    d += x * ad[i];
  }
  srcb[idx] = s;
  dstb[idx] = d;
}

// =====================================================================
// Exact per-row softmax max, computed OUTSIDE the WMMA kernel.
// leaky_relu is monotone, so
//   max_m masked(leaky(src+dst[m])) = leaky(src + max_{adj=1} dst[m]).
// One adj row read serves all 4 heads.  grid: B*N blocks, 128 threads.
// =====================================================================
__global__ __launch_bounds__(128) void rowmax_kernel(
    const float* __restrict__ srcb, const float* __restrict__ dstb,
    const int* __restrict__ adj, float* __restrict__ rmax)
{
  __shared__ float red[128];
  const int bn = blockIdx.x;
  const int b = bn >> 11, n = bn & (NNODES - 1);
  const int tid = threadIdx.x;
  const int* adjrow = adj + ((size_t)b * NNODES + n) * NNODES;

  float mx[NHEAD] = {-3.0e38f, -3.0e38f, -3.0e38f, -3.0e38f};
#pragma unroll
  for (int it = 0; it < NNODES / (128 * 4); ++it) {
    const int m4 = (it * 128 + tid) * 4;
    const int4 a4 = *(const int4*)(adjrow + m4);
#pragma unroll
    for (int h = 0; h < NHEAD; ++h) {
      const float4 d4 =
          *(const float4*)(dstb + (size_t)(b * NHEAD + h) * NNODES + m4);
      if (a4.x) mx[h] = fmaxf(mx[h], d4.x);
      if (a4.y) mx[h] = fmaxf(mx[h], d4.y);
      if (a4.z) mx[h] = fmaxf(mx[h], d4.z);
      if (a4.w) mx[h] = fmaxf(mx[h], d4.w);
    }
  }
  for (int h = 0; h < NHEAD; ++h) {
    red[tid] = mx[h];
    __syncthreads();
    for (int off = 64; off > 0; off >>= 1) {
      if (tid < off) red[tid] = fmaxf(red[tid], red[tid + off]);
      __syncthreads();
    }
    if (tid == 0) {
      float md = red[0];
      float s  = srcb[(size_t)(b * NHEAD + h) * NNODES + n];
      float t  = s + md;
      t = (t >= 0.0f) ? t : ALPHA * t;
      rmax[(size_t)(b * NHEAD + h) * NNODES + n] = (md < -1.0e37f) ? NEGV : t;
    }
    __syncthreads();
  }
}

// =====================================================================
// Fixed-max streaming-softmax GAT attention.  Each wave owns TWO 16-row
// tiles of one (b,head); value-tile B operands are loaded once per m-tile
// and reused by both P tiles (8 WMMA per 8 B128 loads).  With the row max
// precomputed there is NO online rescale: accumulators are only touched
// by WMMA inside the loop (no cross-lane ops, no hazard NOP walls).
// grid: (N/128, B*H), block: 128 (4 waves share the LDS dst stage)
// =====================================================================
__global__ __launch_bounds__(128) void gat_attn_kernel(
    const unsigned short* __restrict__ hT, const float* __restrict__ srcb,
    const float* __restrict__ dstb, const int* __restrict__ adj,
    const float* __restrict__ rmax, float* __restrict__ out)
{
  __shared__ float s_dst[NNODES];     // 8 KB, shared by 4 waves

  const int tid  = threadIdx.x;
  const int wave = tid >> 5;
  const int lane = tid & 31;
  const int half = lane >> 4;
  const int l16  = lane & 15;
  const int bh   = blockIdx.y;
  const int b    = bh / NHEAD;
  const int rowBase = (blockIdx.x * 4 + wave) * 32;

  const float* dstp = dstb + (size_t)bh * NNODES;
  for (int i = tid; i < NNODES; i += 128) s_dst[i] = dstp[i];
  __syncthreads();

  const float srcv0 = srcb[(size_t)bh * NNODES + rowBase + l16];
  const float srcv1 = srcb[(size_t)bh * NNODES + rowBase + 16 + l16];
  const float mx0   = rmax[(size_t)bh * NNODES + rowBase + l16];
  const float mx1   = rmax[(size_t)bh * NNODES + rowBase + 16 + l16];
  const int* adjrow0 = adj + ((size_t)b * NNODES + rowBase + l16) * NNODES;
  const int* adjrow1 = adjrow0 + (size_t)16 * NNODES;
  const unsigned short* hTb = hT + (size_t)bh * DK * NNODES;

  float runsum[2] = {0.0f, 0.0f};
  v8f acc[2][4];
  for (int t = 0; t < 2; ++t)
    for (int s = 0; s < 4; ++s)
      for (int v = 0; v < 8; ++v) acc[t][s][v] = 0.0f;

  for (int m0 = 0; m0 < NNODES; m0 += 32) {
    // ---- issue value-tile B operands early (latency hides under exp) ----
    B16 Bv[4];
#pragma unroll
    for (int s = 0; s < 4; ++s) {
      const uint4* vp = (const uint4*)(hTb + (size_t)(s * 16 + l16) * NNODES +
                                       m0 + 8 * half);
      Bv[s].q[0] = vp[0];
      Bv[s].q[1] = vp[2];
    }
    if (m0 + 32 < NNODES) {
      __builtin_prefetch(adjrow0 + m0 + 32, 0, 0);    // global_prefetch_b8
      __builtin_prefetch(adjrow1 + m0 + 32, 0, 0);
    }

    // ---- dst runs (shared by both row tiles, A-layout K pattern) ----
    const float4* dq  = (const float4*)&s_dst[m0 + 8 * half];
    const float4* dq2 = (const float4*)&s_dst[m0 + 16 + 8 * half];
    float4 dv4[4] = { dq[0], dq[1], dq2[0], dq2[1] };
    const float* dsc = (const float*)dv4;

    B16 pA[2];
#pragma unroll
    for (int t = 0; t < 2; ++t) {
      const int* adjrow = t ? adjrow1 : adjrow0;
      const float srcv  = t ? srcv1 : srcv0;
      const float mx    = t ? mx1 : mx0;
      const int4* aq  = (const int4*)(adjrow + m0 + 8 * half);
      const int4* aq2 = (const int4*)(adjrow + m0 + 16 + 8 * half);
      int4 av4[4] = { aq[0], aq[1], aq2[0], aq2[1] };
      const int* asc = (const int*)av4;

      float lsum = 0.0f;
      for (int e = 0; e < 16; ++e) {
        float x = srcv + dsc[e];
        x = (x >= 0.0f) ? x : ALPHA * x;
        x = (asc[e] == 0) ? NEGV : x;
        float p = __expf(x - mx);
        lsum += p;
        pA[t].s[e] = f2bf(p);
      }
      runsum[t] += lsum;
    }

    // ---- 8 WMMAs: both P tiles consume the same B operands ----
#pragma unroll
    for (int s = 0; s < 4; ++s) {
      acc[0][s] = wmma_bf16(pA[0].v, Bv[s].v, acc[0][s]);
      acc[1][s] = wmma_bf16(pA[1].v, Bv[s].v, acc[1][s]);
    }
  }

  // ---- one cross-lane reduce at the very end ----
  runsum[0] += __shfl_xor(runsum[0], 16, 32);
  runsum[1] += __shfl_xor(runsum[1], 16, 32);

  for (int t = 0; t < 2; ++t) {
    for (int v = 0; v < 8; ++v) {
      float ls = __shfl(runsum[t], v + 8 * half, 32);
      const int orow = rowBase + t * 16 + v + 8 * half;
      for (int s = 0; s < 4; ++s) {
        out[(size_t)bh * NNODES * DK + (size_t)orow * DK + s * 16 + l16] =
            acc[t][s][v] / ls;
      }
    }
  }
}

// =====================================================================
// Block reduction helper
// =====================================================================
__device__ inline float block_sum(float v, float* red, int nthreads) {
  int t = threadIdx.x;
  red[t] = v;
  __syncthreads();
  for (int off = nthreads >> 1; off > 0; off >>= 1) {
    if (t < off) red[t] += red[t + off];
    __syncthreads();
  }
  float r = red[0];
  __syncthreads();
  return r;
}

// x[b,n,:] += elu(layernorm(concat_heads(att[b,:,n,:])))
__global__ __launch_bounds__(256) void ln_elu_res_kernel(
    const float* __restrict__ att, const float* __restrict__ g,
    const float* __restrict__ beta, float* __restrict__ x)
{
  __shared__ float red[256];
  const int bn = blockIdx.x;
  const int b = bn >> 11, n = bn & (NNODES - 1);
  const int f = threadIdx.x;
  const int head = f >> 6, d = f & (DK - 1);

  float v = att[(((size_t)b * NHEAD + head) * NNODES + n) * DK + d];
  float mu  = block_sum(v, red, 256) * (1.0f / HID);
  float sq  = block_sum(v * v, red, 256) * (1.0f / HID);
  float var = sq - mu * mu;
  float y = (v - mu) * rsqrtf(var + 1e-5f) * g[f] + beta[f];
  y = (y > 0.0f) ? y : (__expf(y) - 1.0f);           // ELU
  size_t idx = ((size_t)b * NNODES + n) * HID + f;
  x[idx] = x[idx] + y;
}

// out[b,n,:] = layernorm(mean_heads(att[b,:,n,:]))
__global__ __launch_bounds__(64) void head_mean_ln_kernel(
    const float* __restrict__ att, const float* __restrict__ g,
    const float* __restrict__ beta, float* __restrict__ out)
{
  __shared__ float red[64];
  const int bn = blockIdx.x;
  const int b = bn >> 11, n = bn & (NNODES - 1);
  const int d = threadIdx.x;

  float v = 0.0f;
  for (int head = 0; head < NHEAD; ++head)
    v += att[(((size_t)b * NHEAD + head) * NNODES + n) * DK + d];
  v *= (1.0f / NHEAD);

  float mu  = block_sum(v, red, 64) * (1.0f / DK);
  float sq  = block_sum(v * v, red, 64) * (1.0f / DK);
  float var = sq - mu * mu;
  out[((size_t)b * NNODES + n) * DK + d] =
      (v - mu) * rsqrtf(var + 1e-5f) * g[d] + beta[d];
}

// =====================================================================
extern "C" void kernel_launch(void* const* d_in, const int* in_sizes, int n_in,
                              void* d_out, int out_size, void* d_ws, size_t ws_size,
                              hipStream_t stream) {
  const float* nf   = (const float*)d_in[0];   // (B,N,FIN)
  const int*   adj  = (const int*)  d_in[1];   // (B,N,N)
  const float* Wp   = (const float*)d_in[2];   // (FIN,HID)
  const float* bp   = (const float*)d_in[3];   // (HID)
  const float* W1   = (const float*)d_in[4];   // (H,HID,DK)
  const float* a1   = (const float*)d_in[5];   // (H,2*DK,1)
  const float* l1s  = (const float*)d_in[6];
  const float* l1b  = (const float*)d_in[7];
  const float* W2   = (const float*)d_in[8];   // (H,HID,DK)
  const float* a2   = (const float*)d_in[9];
  const float* l2s  = (const float*)d_in[10];
  const float* l2b  = (const float*)d_in[11];
  float* outp = (float*)d_out;

  const size_t R = (size_t)NB * NNODES;        // 8192 rows

  // ---- workspace carve ----
  float* x    = (float*)d_ws;                  // R*HID           (8 MB)
  float* hbf  = x    + R * HID;                // B*H*N*DK        (8 MB)
  float* att  = hbf  + R * HID;                // B*H*N*DK        (8 MB)
  float* srcb = att  + R * HID;                // B*H*N
  float* dstb = srcb + R * NHEAD;              // B*H*N
  float* rmax = dstb + R * NHEAD;              // B*H*N
  unsigned short* xbf = (unsigned short*)(rmax + R * NHEAD); // R*HID bf16
  unsigned short* hT  = xbf + R * HID;         // B*H*DK*N bf16
  unsigned short* Wt  = hT  + R * HID;         // <= H*DK*HID bf16

  const dim3 blk128(128);
  const int cvt256 = 256;

  // ---- projection: x = nf @ Wp + bp ----
  f32_to_bf16_kernel<<<(int)(R * FIN / 2 + 255) / 256, cvt256, 0, stream>>>(
      nf, (unsigned int*)xbf, (int)(R * FIN / 2));
  w_to_bf16t_kernel<<<dim3((FIN * HID + 255) / 256, 1), cvt256, 0, stream>>>(
      Wp, Wt, FIN, HID);
  gemm_head_kernel<FIN, HID><<<dim3(R / 128, 1, HID / 64), blk128, 0, stream>>>(
      xbf, Wt, bp, x);

  // ---- layer 1 ----
  f32_to_bf16_kernel<<<(int)(R * HID / 2 + 255) / 256, cvt256, 0, stream>>>(
      x, (unsigned int*)xbf, (int)(R * HID / 2));
  w_to_bf16t_kernel<<<dim3((HID * DK + 255) / 256, NHEAD), cvt256, 0, stream>>>(
      W1, Wt, HID, DK);
  gemm_head_kernel<HID, DK><<<dim3(R / 128, NHEAD, 1), blk128, 0, stream>>>(
      xbf, Wt, nullptr, hbf);
  srcdst_kernel<<<(NB * NHEAD * NNODES + 255) / 256, 256, 0, stream>>>(
      hbf, a1, srcb, dstb);
  h_to_bf16t_kernel<<<(int)(R * HID / 256), cvt256, 0, stream>>>(hbf, hT);
  rowmax_kernel<<<NB * NNODES, 128, 0, stream>>>(srcb, dstb, adj, rmax);
  gat_attn_kernel<<<dim3(NNODES / 128, NB * NHEAD), blk128, 0, stream>>>(
      hT, srcb, dstb, adj, rmax, att);
  ln_elu_res_kernel<<<NB * NNODES, 256, 0, stream>>>(att, l1s, l1b, x);

  // ---- layer 2 ----
  f32_to_bf16_kernel<<<(int)(R * HID / 2 + 255) / 256, cvt256, 0, stream>>>(
      x, (unsigned int*)xbf, (int)(R * HID / 2));
  w_to_bf16t_kernel<<<dim3((HID * DK + 255) / 256, NHEAD), cvt256, 0, stream>>>(
      W2, Wt, HID, DK);
  gemm_head_kernel<HID, DK><<<dim3(R / 128, NHEAD, 1), blk128, 0, stream>>>(
      xbf, Wt, nullptr, hbf);
  srcdst_kernel<<<(NB * NHEAD * NNODES + 255) / 256, 256, 0, stream>>>(
      hbf, a2, srcb, dstb);
  h_to_bf16t_kernel<<<(int)(R * HID / 256), cvt256, 0, stream>>>(hbf, hT);
  rowmax_kernel<<<NB * NNODES, 128, 0, stream>>>(srcb, dstb, adj, rmax);
  gat_attn_kernel<<<dim3(NNODES / 128, NB * NHEAD), blk128, 0, stream>>>(
      hT, srcb, dstb, adj, rmax, att);
  head_mean_ln_kernel<<<NB * NNODES, 64, 0, stream>>>(att, l2s, l2b, outp);
}
